// MultiHeadAttention_22634477650582
// MI455X (gfx1250) — matrix operand order repaired and checked by reading, not compiled, not run on hardware
//
#include <hip/hip_runtime.h>
#include <hip/hip_bf16.h>

#define NUM_HEAD 16
#define D_MODEL  1024
#define HEAD_DIM 64
#define SEQ      2048
#define BATCH    2

typedef unsigned short u16;
typedef __attribute__((ext_vector_type(16))) __bf16 v16bf;
typedef __attribute__((ext_vector_type(8)))  __bf16 v8bf;
typedef __attribute__((ext_vector_type(8)))  float  v8f;
typedef __attribute__((ext_vector_type(4)))  int    i32x4;

union Frag16 { v16bf v; v8bf h[2]; };

#if __has_builtin(__builtin_amdgcn_global_load_async_to_lds_b128)
#define HAVE_ASYNC_LDS 1
typedef __attribute__((address_space(1))) i32x4 gas_i32x4;  // global int4
typedef __attribute__((address_space(3))) i32x4 las_i32x4;  // LDS int4
#else
#define HAVE_ASYNC_LDS 0
#endif

// 16-byte global -> LDS copy; async DMA path on gfx1250 when available.
__device__ __forceinline__ void copy16_g2l(u16* dst_lds, const u16* src) {
#if HAVE_ASYNC_LDS
  __builtin_amdgcn_global_load_async_to_lds_b128(
      (gas_i32x4*)(void*)src, (las_i32x4*)(void*)dst_lds, 0, 0);
#else
  *(v8bf*)dst_lds = *(const v8bf*)src;
#endif
}

// wait for outstanding async global->LDS transfers (ASYNCcnt == 0)
__device__ __forceinline__ void async_join() {
#if HAVE_ASYNC_LDS
#if __has_builtin(__builtin_amdgcn_s_wait_asynccnt)
  __builtin_amdgcn_s_wait_asynccnt(0);
#else
  asm volatile("s_wait_asynccnt 0x0" ::: "memory");
#endif
#endif
}

__device__ __forceinline__ u16 f32_to_bf16(float f) {
  unsigned int u = __float_as_uint(f);
  u += 0x7fffu + ((u >> 16) & 1u);   // round-to-nearest-even
  return (u16)(u >> 16);
}

__device__ __forceinline__ v8f vzero8() {
  v8f z;
#pragma unroll
  for (int i = 0; i < 8; ++i) z[i] = 0.0f;
  return z;
}

// ---------------------------------------------------------------------------
// fp32 -> bf16 elementwise convert
// ---------------------------------------------------------------------------
__global__ void cvt_bf16_kernel(const float* __restrict__ in,
                                u16* __restrict__ out, int n) {
  int i = (blockIdx.x * blockDim.x + threadIdx.x) * 4;
  if (i + 3 < n) {
#pragma unroll
    for (int e = 0; e < 4; ++e) out[i + e] = f32_to_bf16(in[i + e]);
  }
}

// fp32 [K][N] -> bf16 transposed [N][K]
__global__ void cvt_bf16_t_kernel(const float* __restrict__ in,
                                  u16* __restrict__ out, int K, int N) {
  int i = blockIdx.x * blockDim.x + threadIdx.x;
  if (i < K * N) {
    int k = i / N, n = i - k * N;
    out[(size_t)n * K + k] = f32_to_bf16(in[i]);
  }
}

// ---------------------------------------------------------------------------
// bf16 GEMM: C[M,N] = A[M,K] * Bt[N,K]^T + bias  (fp32 accumulate via WMMA)
// block tile 128x128x32, 256 threads = 8 waves; each wave owns 32x64.
// Double-buffered LDS; next K-tile staged via async global->LDS DMA while the
// current tile feeds the WMMAs.
// ---------------------------------------------------------------------------
template <bool OUT_BF16>
__global__ void __launch_bounds__(256)
wmma_gemm_kernel(const u16* __restrict__ A, const u16* __restrict__ Bt,
                 const float* __restrict__ bias, void* __restrict__ Cptr,
                 int M, int N, int K) {
  __shared__ __align__(16) u16 As[2][128 * 32];
  __shared__ __align__(16) u16 Bs[2][128 * 32];

  const int m0   = blockIdx.y * 128;
  const int n0   = blockIdx.x * 128;
  const int tid  = threadIdx.x;
  const int wave = tid >> 5;
  const int lane = tid & 31;
  const int lmod = lane & 15;
  const int lhi  = lane >> 4;
  const int wm   = wave >> 1;   // 0..3  (M direction, 32 rows each)
  const int wn   = wave & 1;    // 0..1  (N direction, 64 cols each)

  v8f acc[2][4];
#pragma unroll
  for (int mi = 0; mi < 2; ++mi)
#pragma unroll
    for (int nj = 0; nj < 4; ++nj) acc[mi][nj] = vzero8();

  // cooperative stage of one 128x32 K-tile pair into buffer `buf`
  auto stage = [&](int buf, int k0) {
#pragma unroll
    for (int i = 0; i < 2; ++i) {
      int c   = tid + i * 256;
      int row = c >> 2;
      int cc  = (c & 3) * 8;
      copy16_g2l(&As[buf][row * 32 + cc],
                 A + (size_t)(m0 + row) * K + k0 + cc);
      copy16_g2l(&Bs[buf][row * 32 + cc],
                 Bt + (size_t)(n0 + row) * K + k0 + cc);
    }
  };

  int buf = 0;
  stage(0, 0);
  async_join();
  __syncthreads();

  for (int k0 = 0; k0 < K; k0 += 32) {
    // kick off DMA of the next tile into the shadow buffer
    if (k0 + 32 < K) stage(buf ^ 1, k0 + 32);

    // A fragment: lane holds row m=lane%16, K-chunks [8*lhi..+7] and [+16..]
    Frag16 af[2];
#pragma unroll
    for (int mi = 0; mi < 2; ++mi) {
      int row = wm * 32 + mi * 16 + lmod;
      int kb  = lhi * 8;
      af[mi].h[0] = *(const v8bf*)(&As[buf][row * 32 + kb]);
      af[mi].h[1] = *(const v8bf*)(&As[buf][row * 32 + kb + 16]);
    }
    // B fragment: lane holds col n=lane%16, K-chunks [8*lhi..+7] and [+16..]
    Frag16 bfr[4];
#pragma unroll
    for (int nj = 0; nj < 4; ++nj) {
      int row = wn * 64 + nj * 16 + lmod;
      int kb  = lhi * 8;
      bfr[nj].h[0] = *(const v8bf*)(&Bs[buf][row * 32 + kb]);
      bfr[nj].h[1] = *(const v8bf*)(&Bs[buf][row * 32 + kb + 16]);
    }
#pragma unroll
    for (int mi = 0; mi < 2; ++mi)
#pragma unroll
      for (int nj = 0; nj < 4; ++nj)
        acc[mi][nj] = __builtin_amdgcn_wmma_f32_16x16x32_bf16(
            false, af[mi].v, false, bfr[nj].v, (short)0, acc[mi][nj], false, false);

    async_join();       // shadow-buffer DMA complete (this wave's share)
    __syncthreads();    // everyone done reading buf + done writing buf^1
    buf ^= 1;
  }

  // epilogue: C layout lane=(col), vgpr r => row 8*lhi + r
#pragma unroll
  for (int mi = 0; mi < 2; ++mi) {
#pragma unroll
    for (int nj = 0; nj < 4; ++nj) {
      int   col = n0 + wn * 64 + nj * 16 + lmod;
      float bv  = bias[col];
#pragma unroll
      for (int r = 0; r < 8; ++r) {
        int   row = m0 + wm * 32 + mi * 16 + lhi * 8 + r;
        float v   = acc[mi][nj][r] + bv;
        if (OUT_BF16)
          ((u16*)Cptr)[(size_t)row * N + col] = f32_to_bf16(v);
        else
          ((float*)Cptr)[(size_t)row * N + col] = v;
      }
    }
  }
}

// ---------------------------------------------------------------------------
// Flash-style causal attention. One block per (b, h, 64-query-row block).
// 128 threads = 4 waves; each wave owns 16 query rows.
// qkv (bf16) layout: [b*S+s][3072], head h at col h*192: q|k|v of 64 each.
// ctx written with the reference's [B,H,S,hd].reshape(B,S,1024) index mix.
// ---------------------------------------------------------------------------
__global__ void __launch_bounds__(128)
flash_attn_kernel(const u16* __restrict__ qkv, u16* __restrict__ ctx) {
  const int qblocks = SEQ / 64;               // 32
  const int bh  = blockIdx.x / qblocks;
  const int qb  = blockIdx.x % qblocks;
  const int b   = bh / NUM_HEAD;
  const int h   = bh % NUM_HEAD;
  const int wave = threadIdx.x >> 5;
  const int lane = threadIdx.x & 31;
  const int lmod = lane & 15;
  const int lhi  = lane >> 4;
  const int q0   = qb * 64;

  __shared__ __align__(16) u16 Ks[64 * 64];   // K rows as-is: [key][d]
  __shared__ __align__(16) u16 Vt[64 * 64];   // V transposed: [d][key]
  __shared__ __align__(16) u16 Ps[64 * 64];   // probs bf16:   [row][key]

  const size_t rs = 3 * D_MODEL;              // 3072
  const u16* base = qkv + (size_t)b * SEQ * rs + h * (3 * HEAD_DIM);

  // Q fragments held in registers (2 K-steps of 32 over hd=64)
  Frag16 qf[2];
  {
    int m = q0 + wave * 16 + lmod;
    const u16* qrow = base + (size_t)m * rs;
    int kb = lhi * 8;
    qf[0].h[0] = *(const v8bf*)(qrow + kb);
    qf[0].h[1] = *(const v8bf*)(qrow + kb + 16);
    qf[1].h[0] = *(const v8bf*)(qrow + 32 + kb);
    qf[1].h[1] = *(const v8bf*)(qrow + 48 + kb);
  }

  v8f o[4];
#pragma unroll
  for (int dn = 0; dn < 4; ++dn) o[dn] = vzero8();
  float rmax[8], rsum[8];
#pragma unroll
  for (int r = 0; r < 8; ++r) { rmax[r] = -__builtin_inff(); rsum[r] = 0.0f; }

  for (int j = 0; j <= qb; ++j) {
    const int t0 = j * 64;
    __syncthreads();
    // stage K via async DMA; V through VGPRs (needs transpose): 512 chunks
    for (int c = threadIdx.x; c < 512; c += 128) {
      int key = c >> 3;
      int cc  = (c & 7) * 8;
      const u16* krow = base + (size_t)(t0 + key) * rs + HEAD_DIM;
      copy16_g2l(Ks + key * 64 + cc, krow + cc);
      const u16* vrow = base + (size_t)(t0 + key) * rs + 2 * HEAD_DIM;
      v8bf vv = *(const v8bf*)(vrow + cc);
#pragma unroll
      for (int e = 0; e < 8; ++e)
        Vt[(cc + e) * 64 + key] = ((const u16*)&vv)[e];
    }
    async_join();
    __syncthreads();

    // ---- scores S = Q K^T : M=16 (wave rows), N=64 keys, K=64 dims
    v8f sc[4];
#pragma unroll
    for (int tn = 0; tn < 4; ++tn) sc[tn] = vzero8();
#pragma unroll
    for (int s = 0; s < 2; ++s) {
#pragma unroll
      for (int tn = 0; tn < 4; ++tn) {
        Frag16 bfr;
        int nrow = tn * 16 + lmod;          // key within tile
        int kb   = s * 32 + lhi * 8;        // d
        bfr.h[0] = *(const v8bf*)(Ks + nrow * 64 + kb);
        bfr.h[1] = *(const v8bf*)(Ks + nrow * 64 + kb + 16);
        sc[tn] = __builtin_amdgcn_wmma_f32_16x16x32_bf16(
            false, qf[s].v, false, bfr.v, (short)0, sc[tn], false, false);
      }
    }

    // ---- scale + causal mask (only diagonal block needs masking)
    const float scale = 0.125f;              // 1/sqrt(64)
    const bool  diag  = (j == qb);
#pragma unroll
    for (int tn = 0; tn < 4; ++tn) {
      int key = t0 + tn * 16 + lmod;
#pragma unroll
      for (int r = 0; r < 8; ++r) {
        int   qr = q0 + wave * 16 + lhi * 8 + r;
        float v  = sc[tn][r] * scale;
        sc[tn][r] = (diag && key > qr) ? -__builtin_inff() : v;
      }
    }

    // ---- online softmax (row reductions across 16 lanes)
    float mnew[8], corr[8];
#pragma unroll
    for (int r = 0; r < 8; ++r) {
      float m = fmaxf(fmaxf(sc[0][r], sc[1][r]), fmaxf(sc[2][r], sc[3][r]));
#pragma unroll
      for (int off = 1; off < 16; off <<= 1) m = fmaxf(m, __shfl_xor(m, off, 32));
      mnew[r] = fmaxf(rmax[r], m);
      corr[r] = __expf(rmax[r] - mnew[r]);
      rmax[r] = mnew[r];
    }
#pragma unroll
    for (int r = 0; r < 8; ++r) {
      float psum = 0.0f;
#pragma unroll
      for (int tn = 0; tn < 4; ++tn) {
        float p = __expf(sc[tn][r] - mnew[r]);
        sc[tn][r] = p;
        psum += p;
      }
#pragma unroll
      for (int off = 1; off < 16; off <<= 1) psum += __shfl_xor(psum, off, 32);
      rsum[r] = rsum[r] * corr[r] + psum;
#pragma unroll
      for (int dn = 0; dn < 4; ++dn) o[dn][r] *= corr[r];
    }

    // ---- quantize P to bf16 in LDS (wave-private rows: no barrier needed)
#pragma unroll
    for (int tn = 0; tn < 4; ++tn)
#pragma unroll
      for (int r = 0; r < 8; ++r)
        Ps[(wave * 16 + lhi * 8 + r) * 64 + tn * 16 + lmod] =
            f32_to_bf16(sc[tn][r]);

    // ---- O += P V : M=16 rows, N=64 dims, K=64 keys
#pragma unroll
    for (int s = 0; s < 2; ++s) {
      Frag16 afr;
      int m  = wave * 16 + lmod;
      int kb = s * 32 + lhi * 8;
      afr.h[0] = *(const v8bf*)(Ps + m * 64 + kb);
      afr.h[1] = *(const v8bf*)(Ps + m * 64 + kb + 16);
#pragma unroll
      for (int dn = 0; dn < 4; ++dn) {
        Frag16 bfr;
        int nrow = dn * 16 + lmod;          // d index
        int kb2  = s * 32 + lhi * 8;        // key
        bfr.h[0] = *(const v8bf*)(Vt + nrow * 64 + kb2);
        bfr.h[1] = *(const v8bf*)(Vt + nrow * 64 + kb2 + 16);
        o[dn] = __builtin_amdgcn_wmma_f32_16x16x32_bf16(
            false, afr.v, false, bfr.v, (short)0, o[dn], false, false);
      }
    }
  }

  // ---- normalize + store with reference reshape scramble:
  // ctx[b][h*128 + q/16][(q%16)*64 + d]
#pragma unroll
  for (int r = 0; r < 8; ++r) {
    float inv = 1.0f / rsum[r];
    int   q   = q0 + wave * 16 + lhi * 8 + r;
    int   row = h * 128 + (q >> 4);
    int   cb  = (q & 15) * 64;
    u16* crow = ctx + ((size_t)b * SEQ + row) * D_MODEL + cb;
#pragma unroll
    for (int dn = 0; dn < 4; ++dn)
      crow[dn * 16 + lmod] = f32_to_bf16(o[dn][r] * inv);
  }
}

// ---------------------------------------------------------------------------
// host-side orchestration
// ---------------------------------------------------------------------------
extern "C" void kernel_launch(void* const* d_in, const int* in_sizes, int n_in,
                              void* d_out, int out_size, void* d_ws, size_t ws_size,
                              hipStream_t stream) {
  const float* x     = (const float*)d_in[0];  // [2,2048,1024]
  const float* w_qkv = (const float*)d_in[1];  // [1024,3072]
  const float* b_qkv = (const float*)d_in[2];  // [3072]
  const float* w_out = (const float*)d_in[3];  // [1024,1024]
  const float* b_out = (const float*)d_in[4];  // [1024]
  float* out = (float*)d_out;                  // [2,2048,1024]

  const size_t M = (size_t)BATCH * SEQ;        // 4096

  // workspace layout (bf16 elements)
  u16* xb    = (u16*)d_ws;                       // 4096*1024
  u16* wqkvt = xb    + M * D_MODEL;              // 3072*1024  ([n][k])
  u16* woutt = wqkvt + (size_t)3 * D_MODEL * D_MODEL;   // 1024*1024 ([n][k])
  u16* qkvb  = woutt + (size_t)D_MODEL * D_MODEL;       // 4096*3072
  u16* ctxb  = qkvb  + M * 3 * D_MODEL;          // 4096*1024

  // 1) precision conversion
  {
    int n = (int)(M * D_MODEL);
    cvt_bf16_kernel<<<dim3((n / 4 + 255) / 256), dim3(256), 0, stream>>>(x, xb, n);
    int nq = 3 * D_MODEL * D_MODEL;
    cvt_bf16_t_kernel<<<dim3((nq + 255) / 256), dim3(256), 0, stream>>>(
        w_qkv, wqkvt, D_MODEL, 3 * D_MODEL);
    int no = D_MODEL * D_MODEL;
    cvt_bf16_t_kernel<<<dim3((no + 255) / 256), dim3(256), 0, stream>>>(
        w_out, woutt, D_MODEL, D_MODEL);
  }

  // 2) QKV projection: [4096,1024] x [1024,3072] + b_qkv -> bf16 qkv
  wmma_gemm_kernel<true><<<dim3(3 * D_MODEL / 128, (int)M / 128),
                           dim3(256), 0, stream>>>(
      xb, wqkvt, b_qkv, (void*)qkvb, (int)M, 3 * D_MODEL, D_MODEL);

  // 3) flash attention (causal): 32 (b,h) x 32 query blocks
  flash_attn_kernel<<<dim3(BATCH * NUM_HEAD * (SEQ / 64)), dim3(128), 0, stream>>>(
      qkvb, ctxb);

  // 4) output projection: [4096,1024] x [1024,1024] + b_out -> fp32 d_out
  wmma_gemm_kernel<false><<<dim3(D_MODEL / 128, (int)M / 128),
                            dim3(256), 0, stream>>>(
      ctxb, woutt, b_out, (void*)out, (int)M, D_MODEL, D_MODEL);
}
